// SAGEConv_12884901888281
// MI455X (gfx1250) — matrix operand order repaired and checked by reading, not compiled, not run on hardware
//
#include <hip/hip_runtime.h>
#include <hip/hip_bf16.h>

typedef __attribute__((ext_vector_type(16))) __bf16 v16bf;
typedef __attribute__((ext_vector_type(8)))  float  v8f;

#define DF    128      // feature dim (in == out)
#define TILE  16       // node rows per block / WMMA M
#define LPAD  136      // padded LDS row (bf16 elems): 272B stride -> conflict-free b128 reads
#define LN_EPS 1e-5f

// ---------------------------------------------------------------- zero scratch
__global__ void zero_ws_kernel(float* __restrict__ p, size_t n) {
    size_t i = (size_t)blockIdx.x * blockDim.x + threadIdx.x;
    size_t stride = (size_t)gridDim.x * blockDim.x;
    for (; i < n; i += stride) p[i] = 0.0f;
}

// ------------------------------------------------------- fp32 -> bf16 weights
__global__ void convert_w_kernel(const float* __restrict__ a,
                                 const float* __restrict__ b,
                                 __bf16* __restrict__ oa,
                                 __bf16* __restrict__ ob, int n) {
    int i = blockIdx.x * blockDim.x + threadIdx.x;
    if (i < n) {
        oa[i] = (__bf16)a[i];
        ob[i] = (__bf16)b[i];
    }
}

// ------------------------------------------------- edge scatter (mean aggr.)
// one wave32 per edge: lane l handles dims [4l, 4l+4)
__global__ __launch_bounds__(256)
void sage_scatter_kernel(const float* __restrict__ x,
                         const int* __restrict__ ei,   // [2,E] flat, int32
                         float* __restrict__ agg,
                         float* __restrict__ deg, int E) {
    int wid  = (blockIdx.x * blockDim.x + threadIdx.x) >> 5;
    int lane = threadIdx.x & 31;
    if (wid >= E) return;
    int src = ei[wid];
    int dst = ei[E + wid];
    const float4* xs = (const float4*)(x + (size_t)src * DF);
    float4 v = xs[lane];                       // global_load_b128
    float* ad = agg + (size_t)dst * DF + lane * 4;
    atomicAdd(ad + 0, v.x);                    // global_atomic_add_f32 x4
    atomicAdd(ad + 1, v.y);
    atomicAdd(ad + 2, v.z);
    atomicAdd(ad + 3, v.w);
    if (lane == 0) atomicAdd(deg + dst, 1.0f);
}

// ------------------------------------------------------- WMMA fragment loads
// A fragment (16x32 bf16, M x K), lane = 16*h + m:
//   elems 0..15 -> K = kb + 16h + e
static __device__ __forceinline__ v16bf load_a_frag(const __bf16* row, int kb, int h) {
    union { v16bf v; uint4 q[2]; } u;
    u.q[0] = *(const uint4*)(row + kb + 16 * h);
    u.q[1] = *(const uint4*)(row + kb + 16 * h + 8);
    return u.v;
}
// B fragment (32x16 bf16, K x N), lane = 16*h + n:
//   elems 0..15 -> K = kb + 16h + e, column n  (16 contiguous bf16 of W row)
static __device__ __forceinline__ v16bf load_b_frag(const __bf16* p) {
    union { v16bf v; uint4 q[2]; } u;
    u.q[0] = *(const uint4*)(p);
    u.q[1] = *(const uint4*)(p + 8);
    return u.v;
}

// ------------------------------------- fused mean/GEMM/bias/ReLU/LayerNorm
__global__ __launch_bounds__(256)
void sage_node_kernel(const float* __restrict__ x,
                      const float* __restrict__ agg,
                      const float* __restrict__ deg,
                      const __bf16* __restrict__ Wsbf,   // [128 out][128 in] bf16
                      const __bf16* __restrict__ Wnbf,
                      const float* __restrict__ bias,
                      const float* __restrict__ gamma,
                      const float* __restrict__ beta,
                      float* __restrict__ out, int N) {
    __shared__ __bf16 lx[TILE][LPAD];   // x tile, bf16
    __shared__ __bf16 ln[TILE][LPAD];   // neigh-mean tile, bf16
    __shared__ float  lh[TILE][LPAD];   // post-ReLU tile, fp32

    const int tid   = threadIdx.x;
    const int node0 = blockIdx.x * TILE;

    // ---- stage 16x128 x-tile and neigh-mean-tile into LDS (bf16)
    {
        int r = tid >> 4;            // 0..15
        int c = (tid & 15) * 8;      // 0..120
        int node = node0 + r;
        int nc = node < N ? node : N - 1;           // clamp partial tile
        const float4* xr = (const float4*)(x   + (size_t)nc * DF + c);
        const float4* ar = (const float4*)(agg + (size_t)nc * DF + c);
        float inv = 1.0f / fmaxf(deg[nc], 1.0f);
        float4 x0 = xr[0], x1 = xr[1];
        float4 a0 = ar[0], a1 = ar[1];
        __bf16* px = &lx[r][c];
        __bf16* pn = &ln[r][c];
        px[0] = (__bf16)x0.x; px[1] = (__bf16)x0.y; px[2] = (__bf16)x0.z; px[3] = (__bf16)x0.w;
        px[4] = (__bf16)x1.x; px[5] = (__bf16)x1.y; px[6] = (__bf16)x1.z; px[7] = (__bf16)x1.w;
        pn[0] = (__bf16)(a0.x*inv); pn[1] = (__bf16)(a0.y*inv); pn[2] = (__bf16)(a0.z*inv); pn[3] = (__bf16)(a0.w*inv);
        pn[4] = (__bf16)(a1.x*inv); pn[5] = (__bf16)(a1.y*inv); pn[6] = (__bf16)(a1.z*inv); pn[7] = (__bf16)(a1.w*inv);
    }
    __syncthreads();

    // ---- WMMA: each wave owns a 16-wide output-column slice
    const int wave = tid >> 5;          // 0..7
    const int lane = tid & 31;
    const int m    = lane & 15;         // M for A, N for B/D
    const int h    = lane >> 4;         // lane half
    const int o0   = wave * 16;         // output col base

    v8f acc = {};
    #pragma unroll
    for (int kk = 0; kk < 4; ++kk) {    // self: x @ W_self^T
        int kb = kk * 32;
        v16bf a = load_a_frag(&lx[m][0], kb, h);
        v16bf b = load_b_frag(Wsbf + (size_t)(o0 + m) * DF + kb + 16 * h);
        acc = __builtin_amdgcn_wmma_f32_16x16x32_bf16(false, a, false, b,
                                                      (short)0, acc, false, false);
    }
    #pragma unroll
    for (int kk = 0; kk < 4; ++kk) {    // neigh: mean @ W_neigh^T
        int kb = kk * 32;
        v16bf a = load_a_frag(&ln[m][0], kb, h);
        v16bf b = load_b_frag(Wnbf + (size_t)(o0 + m) * DF + kb + 16 * h);
        acc = __builtin_amdgcn_wmma_f32_16x16x32_bf16(false, a, false, b,
                                                      (short)0, acc, false, false);
    }

    // ---- bias + ReLU into LDS  (D layout: VGPR r -> row r + 8h, col = o0 + m)
    {
        float bi = bias[o0 + m];
        #pragma unroll
        for (int r = 0; r < 8; ++r) {
            float v = acc[r] + bi;
            lh[r + 8 * h][o0 + m] = fmaxf(v, 0.0f);
        }
    }
    __syncthreads();

    // ---- LayerNorm: wave w handles rows 2w, 2w+1; 16 lanes x 8 elems per row
    {
        int row = 2 * wave + h;
        int c   = m * 8;
        float vals[8];
        float s = 0.0f, ss = 0.0f;
        #pragma unroll
        for (int i = 0; i < 8; ++i) {
            float v = lh[row][c + i];
            vals[i] = v;
            s  += v;
            ss += v * v;
        }
        #pragma unroll
        for (int off = 1; off < 16; off <<= 1) {   // reduce within 16-lane group
            s  += __shfl_xor(s,  off, 32);
            ss += __shfl_xor(ss, off, 32);
        }
        float mu   = s * (1.0f / DF);
        float var  = ss * (1.0f / DF) - mu * mu;
        float rstd = rsqrtf(var + LN_EPS);
        int node = node0 + row;
        if (node < N) {
            float* op = out + (size_t)node * DF + c;
            #pragma unroll
            for (int i = 0; i < 8; ++i) {
                int col = c + i;
                op[i] = (vals[i] - mu) * rstd * gamma[col] + beta[col];
            }
        }
    }
}

// --------------------------------------------------------------------- launch
extern "C" void kernel_launch(void* const* d_in, const int* in_sizes, int n_in,
                              void* d_out, int out_size, void* d_ws, size_t ws_size,
                              hipStream_t stream) {
    const float* x     = (const float*)d_in[0];
    const int*   ei    = (const int*)d_in[1];     // [2,E] int
    const float* Ws    = (const float*)d_in[2];
    const float* Wn    = (const float*)d_in[3];
    const float* bias  = (const float*)d_in[4];
    const float* gamma = (const float*)d_in[5];
    const float* beta  = (const float*)d_in[6];
    float* out = (float*)d_out;

    const int N = in_sizes[0] / DF;
    const int E = in_sizes[1] / 2;

    // workspace layout: deg[N] | agg[N*128] | Wsbf[128*128] | Wnbf[128*128]
    float*  deg  = (float*)d_ws;
    float*  agg  = deg + N;
    __bf16* Wsbf = (__bf16*)(agg + (size_t)N * DF);
    __bf16* Wnbf = Wsbf + DF * DF;

    // 1) zero agg + deg (ws is poisoned; must re-zero every call)
    size_t nz = (size_t)N * (DF + 1);
    zero_ws_kernel<<<2048, 256, 0, stream>>>(deg, nz);

    // 2) weights -> bf16
    convert_w_kernel<<<(DF * DF + 255) / 256, 256, 0, stream>>>(Ws, Wn, Wsbf, Wnbf, DF * DF);

    // 3) edge scatter: one wave per edge (8 edges / 256-thread block)
    sage_scatter_kernel<<<(E + 7) / 8, 256, 0, stream>>>(x, ei, agg, deg, E);

    // 4) fused mean / dual-GEMM (WMMA bf16) / bias / ReLU / LayerNorm
    sage_node_kernel<<<(N + TILE - 1) / TILE, 256, 0, stream>>>(
        x, agg, deg, Wsbf, Wnbf, bias, gamma, beta, out, N);
}
